// SelfAttention_68247030334494
// MI455X (gfx1250) — matrix operand order repaired and checked by reading, not compiled, not run on hardware
//
#include <hip/hip_runtime.h>

typedef __bf16 bf16;
typedef __attribute__((ext_vector_type(16))) __bf16 v16bf;
typedef __attribute__((ext_vector_type(8)))  __bf16 bf16x8;
typedef __attribute__((ext_vector_type(8)))  float  v8f;
typedef __attribute__((ext_vector_type(4)))  float  f32x4;
typedef __attribute__((ext_vector_type(4)))  unsigned int u32x4;
typedef __attribute__((ext_vector_type(8)))  unsigned int u32x8;

constexpr int kB  = 4;
constexpr int kC  = 128;
constexpr int kN  = 4096;          // 64*64
constexpr int kD  = 16;            // DQK
constexpr int kDP = 32;            // head dim padded to WMMA K
constexpr int kCN = kC * kN;
constexpr int kSplit = 2;          // key-dimension split across waves
constexpr int kKeysPerHalf = kN / kSplit;

// cross-half-wave exchange (lane <-> lane^16) as a VALU op, no LDS
__device__ __forceinline__ unsigned permx16_u32(unsigned v) {
  return (unsigned)__builtin_amdgcn_permlanex16((int)v, (int)v,
                                                0x76543210, 0xfedcba98,
                                                false, false);
}
__device__ __forceinline__ float permx16_f32(float v) {
  return __int_as_float((int)permx16_u32((unsigned)__float_as_int(v)));
}

// ---------------------------------------------------------------------------
// Kernel 1: Q/K projections.  q[b,d,n] = sum_c Wq[d,c]*x[b,c,n] + bq[d]
// Output layout: Qp/Kp [B][N][32] bf16, dims 16..31 zero-padded (WMMA K=32).
// ---------------------------------------------------------------------------
__global__ void proj_qk_kernel(const float* __restrict__ x,
                               const float* __restrict__ Wq, const float* __restrict__ bq,
                               const float* __restrict__ Wk, const float* __restrict__ bk,
                               bf16* __restrict__ Qp, bf16* __restrict__ Kp) {
  int tid = blockIdx.x * blockDim.x + threadIdx.x;
  if (tid >= kB * kN) return;
  int n = tid & (kN - 1);
  int b = tid >> 12;
  const float* xp = x + (size_t)b * kCN + n;
  float qa[kD], ka[kD];
#pragma unroll
  for (int d = 0; d < kD; ++d) { qa[d] = bq[d]; ka[d] = bk[d]; }
  for (int c = 0; c < kC; ++c) {
    float xv = xp[(size_t)c * kN];
#pragma unroll
    for (int d = 0; d < kD; ++d) {
      qa[d] = fmaf(Wq[d * kC + c], xv, qa[d]);
      ka[d] = fmaf(Wk[d * kC + c], xv, ka[d]);
    }
  }
  bf16* q = Qp + (size_t)tid * kDP;
  bf16* k = Kp + (size_t)tid * kDP;
#pragma unroll
  for (int d = 0; d < kD; ++d) {
    q[d]      = (bf16)qa[d];
    k[d]      = (bf16)ka[d];
    q[kD + d] = (bf16)0.0f;
    k[kD + d] = (bf16)0.0f;
  }
}

// ---------------------------------------------------------------------------
// Kernel 2: V projection -> Vt [B][C][N] bf16 (channel-major: keys contiguous)
// ---------------------------------------------------------------------------
__global__ void proj_v_kernel(const float* __restrict__ x,
                              const float* __restrict__ Wv, const float* __restrict__ bv,
                              bf16* __restrict__ Vt) {
  int tid = blockIdx.x * blockDim.x + threadIdx.x;
  if (tid >= kB * 8 * kN) return;
  int n  = tid & (kN - 1);
  int cb = (tid >> 12) & 7;
  int b  = tid >> 15;
  const float* xp = x + (size_t)b * kCN + n;
  float acc[16];
#pragma unroll
  for (int i = 0; i < 16; ++i) acc[i] = bv[cb * 16 + i];
  for (int c = 0; c < kC; ++c) {
    float xv = xp[(size_t)c * kN];
#pragma unroll
    for (int i = 0; i < 16; ++i)
      acc[i] = fmaf(Wv[(cb * 16 + i) * kC + c], xv, acc[i]);
  }
#pragma unroll
  for (int i = 0; i < 16; ++i)
    Vt[((size_t)b * kC + cb * 16 + i) * kN + n] = (bf16)acc[i];
}

// ---------------------------------------------------------------------------
// Kernel 3: transposed-score flash attention with 2-way key split.
// Block = 4 waves = 2 query tiles x 2 key halves -> 2048 waves total.
// Per-wave loop over its 2048 keys in 32-key steps:
//   2x v_wmma  S^T = K_tile[A] x Q^T[B]   (lane holds a query COLUMN)
//   online softmax: in-lane max/sum + 1 permlanex16 exchange (no LDS in loop)
//   P^T B-operand assembled with 4 permlanex16 (packed bf16) + selects
//   8x v_wmma  O^T += Vt_tile[A] x P^T[B]
// End: key-half 1 publishes (O,m,l) to LDS; key-half 0 merges with
// exp(m_h - m_final) corrections and does the fused residual epilogue.
// ---------------------------------------------------------------------------
__global__ __launch_bounds__(128) void
attn_kernel(const bf16* __restrict__ Qp, const bf16* __restrict__ Kp,
            const bf16* __restrict__ Vt, const float* __restrict__ x,
            const float* __restrict__ gamma_p, float* __restrict__ out) {
  __shared__ __align__(32) float ldsO[2][8 * 32 * 8];   // [qt][cg*32+lane][v]
  __shared__ float ldsM[2][32];
  __shared__ float ldsL[2][32];

  const int lane = threadIdx.x & 31;
  const int wave = threadIdx.x >> 5;
  const int qt   = wave & 1;                    // query tile within block
  const int half = wave >> 1;                   // key half (0/1)
  const int tile = blockIdx.x * 2 + qt;         // 1024 query tiles total
  const int b    = tile >> 8;                   // 256 query tiles per batch
  const int qb   = (tile & 255) * 16;           // first query row of tile
  const int g    = lane >> 4;                   // half-wave group (0/1)
  const int j    = lane & 15;
  const int kb0  = half * kKeysPerHalf;

  // ---- Q^T fragment (B operand, 32x16): lane holds column (query qb+j),
  //      k-dims 8g..8g+7 and 16+8g..16+8g+7 of the padded Qp row.
  v16bf qf;
  {
    const bf16* qrow = Qp + ((size_t)b * kN + qb + j) * kDP;
    bf16x8 q0 = *(const bf16x8*)(qrow + 8 * g);
    bf16x8 q1 = *(const bf16x8*)(qrow + 16 + 8 * g);
#pragma unroll
    for (int e = 0; e < 8; ++e) { qf[e] = q0[e]; qf[8 + e] = q1[e]; }
  }

  v8f O[8];                                     // O^T accumulators (8 ch-groups)
#pragma unroll
  for (int cg = 0; cg < 8; ++cg) O[cg] = (v8f){0.f,0.f,0.f,0.f,0.f,0.f,0.f,0.f};
  float m = -3.0e38f, l = 0.0f;                 // per-query (per-lane) scalars
  const v8f czero = (v8f){0.f,0.f,0.f,0.f,0.f,0.f,0.f,0.f};

  for (int kk = 0; kk < kKeysPerHalf; kk += 32) {
    const int kb = kb0 + kk;
    // ---- K A-operands for two 16-key tiles
    v16bf kf0, kf1;
    {
      const bf16* kr0 = Kp + ((size_t)b * kN + kb + j) * kDP;
      const bf16* kr1 = kr0 + 16 * kDP;
      bf16x8 a0 = *(const bf16x8*)(kr0 + 8 * g);
      bf16x8 a1 = *(const bf16x8*)(kr0 + 16 + 8 * g);
      bf16x8 b0 = *(const bf16x8*)(kr1 + 8 * g);
      bf16x8 b1 = *(const bf16x8*)(kr1 + 16 + 8 * g);
#pragma unroll
      for (int e = 0; e < 8; ++e) {
        kf0[e] = a0[e]; kf0[8 + e] = a1[e];
        kf1[e] = b0[e]; kf1[8 + e] = b1[e];
      }
    }
    // S^T tiles: rows = keys (v + 8g), column = query j
    v8f S0 = __builtin_amdgcn_wmma_f32_16x16x32_bf16(false, kf0, false, qf,
                                                     (short)0, czero, false, false);
    v8f S1 = __builtin_amdgcn_wmma_f32_16x16x32_bf16(false, kf1, false, qf,
                                                     (short)0, czero, false, false);

    // ---- online softmax over keys: in-lane + one cross-half exchange
    float smax = S0[0];
#pragma unroll
    for (int v = 1; v < 8; ++v) smax = fmaxf(smax, S0[v]);
#pragma unroll
    for (int v = 0; v < 8; ++v) smax = fmaxf(smax, S1[v]);
    smax = fmaxf(smax, permx16_f32(smax));
    float mnew = fmaxf(m, smax);
    float corr = __expf(m - mnew);
    m = mnew;
    float p0[8], p1[8], rs = 0.0f;
#pragma unroll
    for (int v = 0; v < 8; ++v) { p0[v] = __expf(S0[v] - mnew); rs += p0[v]; }
#pragma unroll
    for (int v = 0; v < 8; ++v) { p1[v] = __expf(S1[v] - mnew); rs += p1[v]; }
    rs += permx16_f32(rs);
    l = l * corr + rs;
#pragma unroll
    for (int cg = 0; cg < 8; ++cg)
#pragma unroll
      for (int v = 0; v < 8; ++v) O[cg][v] *= corr;

    // ---- build P^T B-operand (32 keys x 16 queries), column = query j.
    bf16x8 pk0, pk1;
#pragma unroll
    for (int v = 0; v < 8; ++v) { pk0[v] = (bf16)p0[v]; pk1[v] = (bf16)p1[v]; }
    u32x4 own0 = __builtin_bit_cast(u32x4, pk0);
    u32x4 own1 = __builtin_bit_cast(u32x4, pk1);
    u32x4 send, recv;
#pragma unroll
    for (int i = 0; i < 4; ++i) send[i] = g ? own0[i] : own1[i];
#pragma unroll
    for (int i = 0; i < 4; ++i) recv[i] = permx16_u32(send[i]);
    u32x8 pd;
#pragma unroll
    for (int i = 0; i < 4; ++i) {
      pd[i]     = g ? recv[i] : own0[i];
      pd[4 + i] = g ? own1[i] : recv[i];
    }
    v16bf pf = __builtin_bit_cast(v16bf, pd);

    // ---- O^T += Vt_tile (A: 16ch x 32keys) x P^T (B: 32keys x 16q)
#pragma unroll
    for (int cg = 0; cg < 8; ++cg) {
      const bf16* vr = Vt + ((size_t)b * kC + cg * 16 + j) * kN + kb;
      bf16x8 v0 = *(const bf16x8*)(vr + 16 * g);
      bf16x8 v1 = *(const bf16x8*)(vr + 16 * g + 8);
      v16bf vf;
#pragma unroll
      for (int e = 0; e < 8; ++e) { vf[e] = v0[e]; vf[8 + e] = v1[e]; }
      O[cg] = __builtin_amdgcn_wmma_f32_16x16x32_bf16(false, vf, false, pf,
                                                      (short)0, O[cg], false, false);
    }
  }

  // ---- split-K merge: key-half 1 publishes, key-half 0 combines + stores
  if (half == 1) {
#pragma unroll
    for (int cg = 0; cg < 8; ++cg)
      *(v8f*)&ldsO[qt][(cg * 32 + lane) * 8] = O[cg];
    ldsM[qt][lane] = m;
    ldsL[qt][lane] = l;
  }
  __syncthreads();
  if (half == 0) {
    const float m1 = ldsM[qt][lane];
    const float l1 = ldsL[qt][lane];
    const float mf = fmaxf(m, m1);
    const float a0 = __expf(m - mf);
    const float a1 = __expf(m1 - mf);
    const float lf = l * a0 + l1 * a1;
    const float scl = gamma_p[0] / lf;
    const float s0 = a0 * scl;
    const float s1 = a1 * scl;
    // O^T row c = cg*16 + v + 8g, column query n = qb+j; flat-in-batch index
    // n*128 + ch is shared with x -> vectorized fused residual stores.
#pragma unroll
    for (int cg = 0; cg < 8; ++cg) {
      v8f o1 = *(const v8f*)&ldsO[qt][(cg * 32 + lane) * 8];
      size_t base = (size_t)b * kCN + (size_t)(qb + j) * kC + cg * 16 + 8 * g;
      f32x4 xa = *(const f32x4*)(x + base);
      f32x4 xb = *(const f32x4*)(x + base + 4);
      f32x4 oa, ob;
#pragma unroll
      for (int i = 0; i < 4; ++i) {
        oa[i] = O[cg][i]     * s0 + o1[i]     * s1 + xa[i];
        ob[i] = O[cg][4 + i] * s0 + o1[4 + i] * s1 + xb[i];
      }
      *(f32x4*)(out + base)     = oa;
      *(f32x4*)(out + base + 4) = ob;
    }
  }
}

// ---------------------------------------------------------------------------
extern "C" void kernel_launch(void* const* d_in, const int* in_sizes, int n_in,
                              void* d_out, int out_size, void* d_ws, size_t ws_size,
                              hipStream_t stream) {
  const float* x     = (const float*)d_in[0];
  const float* Wq    = (const float*)d_in[1];
  const float* bq    = (const float*)d_in[2];
  const float* Wk    = (const float*)d_in[3];
  const float* bk    = (const float*)d_in[4];
  const float* Wv    = (const float*)d_in[5];
  const float* bv    = (const float*)d_in[6];
  const float* gamma = (const float*)d_in[7];
  float* out = (float*)d_out;

  // workspace: Qp 1MB + Kp 1MB + Vt 4MB  (bf16)
  bf16* Qp = (bf16*)d_ws;
  bf16* Kp = Qp + (size_t)kB * kN * kDP;
  bf16* Vt = Kp + (size_t)kB * kN * kDP;

  proj_qk_kernel<<<(kB * kN) / 256, 256, 0, stream>>>(x, Wq, bq, Wk, bk, Qp, Kp);
  proj_v_kernel<<<(kB * 8 * kN) / 256, 256, 0, stream>>>(x, Wv, bv, Vt);
  // 1024 query tiles x 2 key halves = 2048 waves; 4 waves per block
  attn_kernel<<<512, 128, 0, stream>>>(Qp, Kp, Vt, x, gamma, out);
}
